// GaussianShaper_4157528343031
// MI455X (gfx1250) — matrix operand order repaired and checked by reading, not compiled, not run on hardware
//
#include <hip/hip_runtime.h>

typedef __attribute__((ext_vector_type(16))) __bf16          v16bf;
typedef __attribute__((ext_vector_type(8)))  float           v8f;
typedef __attribute__((ext_vector_type(16))) unsigned short  u16x16;
typedef __attribute__((ext_vector_type(8)))  unsigned short  u16x8;
typedef __attribute__((ext_vector_type(4)))  unsigned int    u32x4;
typedef __attribute__((ext_vector_type(8)))  int             i32x8;
typedef __attribute__((ext_vector_type(4)))  int             i32x4;

#define HISTORY 64
#define WINDOW  32
#define DIM     32
#define NPAT    16
#define NB      128
#define LL      131072
#define TT      4096

#if defined(__gfx1250__) && __has_builtin(__builtin_amdgcn_tensor_load_to_lds) && \
    __has_builtin(__builtin_amdgcn_s_wait_tensorcnt)
#define HAVE_TDM 1
#else
#define HAVE_TDM 0
#endif

#if defined(__gfx1250__) && __has_builtin(__builtin_amdgcn_global_load_async_to_lds_b128) && \
    __has_builtin(__builtin_amdgcn_s_wait_asynccnt)
#define HAVE_ASYNC 1
#else
#define HAVE_ASYNC 0
#endif

__device__ __forceinline__ unsigned short f2bf(float f) {
    unsigned int u = __float_as_uint(f);
    u += 0x7FFFu + ((u >> 16) & 1u);            // round-to-nearest-even
    return (unsigned short)(u >> 16);
}
__device__ __forceinline__ v16bf as_bf16(u16x16 u) {
    return __builtin_bit_cast(v16bf, u);
}
__device__ __forceinline__ u16x16 cat8(u16x8 lo, u16x8 hi) {
    return __builtin_shufflevector(lo, hi, 0,1,2,3,4,5,6,7,8,9,10,11,12,13,14,15);
}
// pack 16 f32 (A-layout: off..off+7, off+16..off+23) from LDS into bf16 vector
__device__ __forceinline__ u16x16 packA(const float* s, int off) {
    float4 f0 = *(const float4*)(s + off);
    float4 f1 = *(const float4*)(s + off + 4);
    float4 f2 = *(const float4*)(s + off + 16);
    float4 f3 = *(const float4*)(s + off + 20);
    u16x16 v;
    v[0]=f2bf(f0.x);  v[1]=f2bf(f0.y);  v[2]=f2bf(f0.z);  v[3]=f2bf(f0.w);
    v[4]=f2bf(f1.x);  v[5]=f2bf(f1.y);  v[6]=f2bf(f1.z);  v[7]=f2bf(f1.w);
    v[8]=f2bf(f2.x);  v[9]=f2bf(f2.y);  v[10]=f2bf(f2.z); v[11]=f2bf(f2.w);
    v[12]=f2bf(f3.x); v[13]=f2bf(f3.y); v[14]=f2bf(f3.z); v[15]=f2bf(f3.w);
    return v;
}

#if HAVE_ASYNC
typedef __attribute__((address_space(1))) i32x4 g_i32x4;     // global-AS 128b payload
typedef __attribute__((address_space(3))) i32x4 l_i32x4;     // LDS-AS 128b payload
__device__ __forceinline__ void async_cp128(const float* g, float* l) {
    __builtin_amdgcn_global_load_async_to_lds_b128(
        (g_i32x4*)g, (l_i32x4*)l, 0, 0);
}
#endif

// ---------------------------------------------------------------------------
// Kernel 1: fused conv(stride32,K=64) -> relu+bias -> @keys -> clip => scores
// ONE wave per workgroup (descriptor values uniform from blockIdx).
// x window staged by the Tensor Data Mover; WMMA bf16 16x16x32.
// ---------------------------------------------------------------------------
__global__ __launch_bounds__(32) void k_scores(
    const float* __restrict__ x,      // (B, L)
    const float* __restrict__ conv_w, // (DIM, 1, HISTORY)
    const float* __restrict__ conv_b, // (DIM,)
    const float* __restrict__ keys,   // (DIM, NPAT)
    float* __restrict__ scores)       // (B, T, NPAT)
{
    __shared__ __align__(16) float          chunkF[544]; // x window, f32
    __shared__ __align__(16) unsigned short hmat[512];   // 16x32 hidden, bf16

    const int lane = threadIdx.x & 31;
    const int W    = blockIdx.x;              // 32768 wave-tiles
    const int b    = W >> 8;                  // / (T/16)
    const int t0   = (W & 255) << 4;          // tile start in t

    const long  rowBase = (long)t0 * WINDOW - (HISTORY - 1);
    const float* xb = x + (long)b * LL;
    const int   pad = (t0 == 0) ? (HISTORY - 1) : 0;     // leading zeros at t0==0
    const long  start = rowBase + pad;                   // >= 0
    const int   nelem = 544 - pad;

#if HAVE_TDM
    if (pad) {                                  // zero-fill the padded head
        for (int i = lane; i < pad; i += 32) chunkF[i] = 0.0f;
    }
    {
        // ---- Tensor DMA: 1-D tile of nelem dwords  global -> LDS ----
        const unsigned long long ga = (unsigned long long)(const void*)(xb + start);
        const unsigned lds = (unsigned)(unsigned long long)(const void*)chunkF + (unsigned)pad * 4u;
        u32x4 g0;
        g0[0] = 1u;                                             // count=1, user mode
        g0[1] = lds;                                            // lds_addr (bytes)
        g0[2] = (unsigned)(ga & 0xFFFFFFFFull);                 // global_addr[31:0]
        g0[3] = (unsigned)((ga >> 32) & 0x01FFFFFFull) | (2u << 30); // addr[56:32] | type=2
        const unsigned td0 = (unsigned)nelem;                   // tensor_dim0
        const unsigned tl0 = (unsigned)nelem;                   // tile_dim0
        i32x8 g1;
        g1[0] = (int)(2u << 16);                                // data_size=4B
        g1[1] = (int)((td0 & 0xFFFFu) << 16);                   // tensor_dim0 lo16
        g1[2] = (int)(((td0 >> 16) & 0xFFFFu) | (1u << 16));    // tensor_dim0 hi16 | tensor_dim1=1 lo16
        g1[3] = (int)((tl0 & 0xFFFFu) << 16);                   // tensor_dim1 hi16=0 | tile_dim0
        g1[4] = 0;                                              // tile_dim1=0 (1-D), tile_dim2=0
        g1[5] = (int)td0;                                       // tensor_dim0_stride lo32
        g1[6] = 0;                                              // stride hi16 | dim1_stride lo16
        g1[7] = 0;
        i32x4 gz4 = {0, 0, 0, 0};
        i32x8 gz8 = {0, 0, 0, 0, 0, 0, 0, 0};
        __builtin_amdgcn_tensor_load_to_lds(g0, g1, gz4, gz4, gz8, 0);
    }
#else
    #pragma unroll
    for (int j = 0; j < 17; ++j) {              // fallback: manual staging
        int idx = lane + 32 * j;                // 0..543
        long g = rowBase + idx;
        chunkF[idx] = (g >= 0) ? xb[g] : 0.0f;
    }
#endif

    // ---- B matrices (per-lane gathers into WMMA B layout) while DMA runs ----
    const int n  = lane & 15;
    const int hi = lane >> 4;
    u16x16 bw[2][2];   // [n-tile: d 0-15 / 16-31][k-step: h 0-31 / 32-63]
    #pragma unroll
    for (int nt = 0; nt < 2; ++nt) {
        const int d = n + 16 * nt;
        #pragma unroll
        for (int ks = 0; ks < 2; ++ks) {
            u16x16 tv;
            #pragma unroll
            for (int j = 0; j < 8; ++j) {
                int k0 = ks * 32 + hi * 8 + 2 * j + (j >= 4 ? 8 : 0);
                tv[2*j]   = f2bf(conv_w[d * HISTORY + k0]);
                tv[2*j+1] = f2bf(conv_w[d * HISTORY + k0 + 1]);
            }
            bw[nt][ks] = tv;
        }
    }
    u16x16 bk;  // keys as B (K=32 dims, N=16 patterns)
    #pragma unroll
    for (int j = 0; j < 8; ++j) {
        int k0 = hi * 8 + 2 * j + (j >= 4 ? 8 : 0);
        bk[2*j]   = f2bf(keys[k0 * NPAT + n]);
        bk[2*j+1] = f2bf(keys[(k0 + 1) * NPAT + n]);
    }
    const float bias0 = conv_b[n];
    const float bias1 = conv_b[n + 16];

#if HAVE_TDM
    __builtin_amdgcn_s_wait_tensorcnt(0);
#endif
    __syncthreads();

    // ---- A tiles: f32 b128 loads from LDS, convert to bf16 A layout ----
    const int r    = lane & 15;
    const int base = r * 32 + hi * 8;
    v16bf A0 = as_bf16(packA(chunkF, base));
    v16bf A1 = as_bf16(packA(chunkF, base + 32));

    // ---- GEMM1: (16x64)x(64x32), two n-tiles, two k-steps each ----
    v8f c0 = {};
    v8f c1 = {};
    c0 = __builtin_amdgcn_wmma_f32_16x16x32_bf16(false, A0, false, as_bf16(bw[0][0]),
                                                 (short)0, c0, false, false);
    c0 = __builtin_amdgcn_wmma_f32_16x16x32_bf16(false, A1, false, as_bf16(bw[0][1]),
                                                 (short)0, c0, false, false);
    c1 = __builtin_amdgcn_wmma_f32_16x16x32_bf16(false, A0, false, as_bf16(bw[1][0]),
                                                 (short)0, c1, false, false);
    c1 = __builtin_amdgcn_wmma_f32_16x16x32_bf16(false, A1, false, as_bf16(bw[1][1]),
                                                 (short)0, c1, false, false);

    // ---- bias + relu, stage hidden 16x32 row-major bf16 in LDS ----
    #pragma unroll
    for (int i = 0; i < 8; ++i) {
        int m = i + 8 * hi;
        hmat[m * 32 + n]      = f2bf(fmaxf(c0[i] + bias0, 0.0f));
        hmat[m * 32 + n + 16] = f2bf(fmaxf(c1[i] + bias1, 0.0f));
    }
    __syncthreads();

    // ---- GEMM2: (16x32)x(32x16) with keys ----
    u16x8 hlo = *(const u16x8*)(hmat + base);
    u16x8 hhi = *(const u16x8*)(hmat + base + 16);
    v16bf A2 = as_bf16(cat8(hlo, hhi));
    v8f c2 = {};
    c2 = __builtin_amdgcn_wmma_f32_16x16x32_bf16(false, A2, false, as_bf16(bk),
                                                 (short)0, c2, false, false);

    // ---- clip to [0,6] and store scores[b][t][p] ----
    float* sb = scores + ((long)b * TT + t0) * NPAT + n;
    #pragma unroll
    for (int i = 0; i < 8; ++i) {
        float s = fminf(fmaxf(c2[i], 0.0f), 6.0f);
        sb[(i + 8 * hi) * NPAT] = s;
    }
}

// ---------------------------------------------------------------------------
// Kernel 2: sequential softmax scan over T.  16 lanes = 16 patterns,
// 2 batches per wave32.  Async double-buffered score tiles in LDS.
// ---------------------------------------------------------------------------
__device__ __forceinline__ void scan_step(float s, float& avg, float s0, float s1,
                                          int p, float2* obt) {
    float e = __expf(s - avg);               // scores in [0,6]: no max-sub needed
    float sum = e;
    sum += __shfl_xor(sum, 1);
    sum += __shfl_xor(sum, 2);
    sum += __shfl_xor(sum, 4);
    sum += __shfl_xor(sum, 8);
    float prob = e * __builtin_amdgcn_rcpf(sum);
    avg += prob - 0.0625f;
    float o = prob * s0;
    float d = prob * s1;
    o += __shfl_xor(o, 1); o += __shfl_xor(o, 2);
    o += __shfl_xor(o, 4); o += __shfl_xor(o, 8);
    d += __shfl_xor(d, 1); d += __shfl_xor(d, 2);
    d += __shfl_xor(d, 4); d += __shfl_xor(d, 8);
    if (p == 0) *obt = make_float2(o, d);
}

#define SCH 16           // t-steps per async chunk
#define SCAN_WAVES 4

__global__ __launch_bounds__(128) void k_scan(
    const float* __restrict__ scores,   // (B, T, NPAT)
    const float* __restrict__ avg0,     // (B, NPAT)
    const float* __restrict__ shapes,   // (NPAT, 2)
    float* __restrict__ osbuf)          // (B, T, 2)
{
#if HAVE_ASYNC
    __shared__ __align__(16) float slds[SCAN_WAVES * 1024]; // 2 bufs x 512 f32 per wave
#endif
    const int wv   = threadIdx.x >> 5;
    const int lane = threadIdx.x & 31;
    const int wid  = blockIdx.x * SCAN_WAVES + wv;
    const int hi   = lane >> 4;
    const int p    = lane & 15;
    const int b    = wid * 2 + hi;

    float avg = avg0[b * NPAT + p];
    const float s0 = shapes[p * 2 + 0];
    const float s1 = shapes[p * 2 + 1];
    float2* ob = (float2*)osbuf + (long)b * TT;

#if HAVE_ASYNC
    const float* g0 = scores + (long)(wid * 2)     * TT * NPAT;
    const float* g1 = scores + (long)(wid * 2 + 1) * TT * NPAT;
    float* lw = slds + wv * 1024;
    auto issue = [&](int c, int u) {
        const float* a0 = g0 + (long)c * SCH * NPAT;  // 256 f32 per batch per chunk
        const float* a1 = g1 + (long)c * SCH * NPAT;
        float* dst = lw + u * 512;
        async_cp128(a0 + lane * 4,       dst + lane * 4);
        async_cp128(a0 + 128 + lane * 4, dst + 128 + lane * 4);
        async_cp128(a1 + lane * 4,       dst + 256 + lane * 4);
        async_cp128(a1 + 128 + lane * 4, dst + 384 + lane * 4);
    };
    issue(0, 0);
    const int NCH = TT / SCH;
    int t = 0;
    for (int c = 0; c < NCH; ++c) {
        const int u = c & 1;
        if (c + 1 < NCH) {
            issue(c + 1, u ^ 1);
            __builtin_amdgcn_s_wait_asynccnt(4);   // chunk c done (in-order), next in flight
        } else {
            __builtin_amdgcn_s_wait_asynccnt(0);
        }
        const float* lb = lw + u * 512 + hi * 256 + p;
        #pragma unroll 4
        for (int j = 0; j < SCH; ++j, ++t) {
            scan_step(lb[j * 16], avg, s0, s1, p, ob + t);
        }
    }
#else
    const float* sb = scores + (long)b * TT * NPAT + p;
    for (int t = 0; t < TT; ++t) {
        if ((t & 31) == 0)
            __builtin_prefetch(sb + (long)(t + 64) * NPAT, 0, 1);
        scan_step(sb[(long)t * NPAT], avg, s0, s1, p, ob + t);
    }
#endif
}

// ---------------------------------------------------------------------------
// Kernel 3: out = relu(offset + noise*std - x), float4-vectorized stream.
// ---------------------------------------------------------------------------
__global__ __launch_bounds__(256) void k_out(
    const float* __restrict__ x,
    const float* __restrict__ noise,
    const float* __restrict__ osbuf,
    float* __restrict__ out)
{
    const long g4 = (long)blockIdx.x * blockDim.x + threadIdx.x; // quad index
    const long g  = g4 * 4;
    const int b = (int)(g >> 17);        // / L
    const int l = (int)(g & (LL - 1));
    const int t = l >> 5;

    float2 os = ((const float2*)osbuf)[(long)b * TT + t];
    float4 nz = ((const float4*)noise)[g4];
    float4 xx = ((const float4*)x)[g4];
    float4 r;
    r.x = fmaxf(os.x + nz.x * os.y - xx.x, 0.0f);
    r.y = fmaxf(os.x + nz.y * os.y - xx.y, 0.0f);
    r.z = fmaxf(os.x + nz.z * os.y - xx.z, 0.0f);
    r.w = fmaxf(os.x + nz.w * os.y - xx.w, 0.0f);
    ((float4*)out)[g4] = r;
}

// ---------------------------------------------------------------------------
extern "C" void kernel_launch(void* const* d_in, const int* in_sizes, int n_in,
                              void* d_out, int out_size, void* d_ws, size_t ws_size,
                              hipStream_t stream) {
    const float* x      = (const float*)d_in[0];
    const float* avgsc  = (const float*)d_in[1];
    const float* noise  = (const float*)d_in[2];
    const float* conv_w = (const float*)d_in[3];
    const float* conv_b = (const float*)d_in[4];
    const float* keys   = (const float*)d_in[5];
    const float* shapes = (const float*)d_in[6];
    float* out = (float*)d_out;

    float* scores = (float*)d_ws;                          // 128*4096*16 f32 = 32 MB
    float* osbuf  = scores + (size_t)NB * TT * NPAT;       // 128*4096*2  f32 =  4 MB

    // 32768 single-wave tiles of 16 t-rows (uniform TDM descriptors)
    k_scores<<<32768, 32, 0, stream>>>(x, conv_w, conv_b, keys, scores);
    // 64 waves = 128 batches, 2 per wave
    k_scan<<<16, 128, 0, stream>>>(scores, avgsc, shapes, osbuf);
    // 16.7M outputs, 4 per thread
    k_out<<<16384, 256, 0, stream>>>(x, noise, osbuf, out);
}
